// myVMLMF_CELL_26680336843034
// MI455X (gfx1250) — matrix operand order repaired and checked by reading, not compiled, not run on hardware
//
#include <hip/hip_runtime.h>
#include <cstdint>
#include <cstddef>

// ---------------------------------------------------------------------------
// VMLMF LSTM cell, fused bf16-WMMA implementation for gfx1250 (MI455X).
//   B=4096, I=512, H=1024, rank WR=UR=128
// Phase 0: f32->bf16 conversions (x,h,V_x,V_h), transpose-convert U_x/U_h,
//          corr_x[4,I], corr_h[4,H] reductions.
// Phase 1: Tx = x@U_x, Th = h@U_h via v_wmma_f32_16x16x32_bf16 (bf16 out).
// Phase 2: per 16x16 tile, acc_g = Tx@Vx^T + Th@Vh^T for all 4 gates (32
//          WMMAs), then fused LSTM epilogue -> h_next, c_next. The [B,4H]
//          "lowered" intermediates are never materialized.
// ---------------------------------------------------------------------------

#define BDIM 4096
#define IDIM 512
#define HDIM 1024
#define RNK  128

typedef __attribute__((ext_vector_type(16))) __bf16 v16bf;
typedef __attribute__((ext_vector_type(8)))  float  v8f;

union FragU {
    uint4  q[2];
    v16bf  v;
};

// A-matrix 16x32 bf16 fragment (ISA 7.12.2 layout):
//   lanes 0-15  : row m = lane,   v0-3 = K 0..7,   v4-7 = K 16..23
//   lanes 16-31 : row m = lane-16,v0-3 = K 8..15,  v4-7 = K 24..31
__device__ inline v16bf load_a_frag(const __bf16* A, int lda, int row0, int k0, int lane) {
    int r  = lane & 15;
    int hf = (lane >> 4) & 1;
    const __bf16* p = A + (size_t)(row0 + r) * lda + k0 + hf * 8;
    FragU f;
    f.q[0] = *reinterpret_cast<const uint4*>(p);        // 8 bf16
    f.q[1] = *reinterpret_cast<const uint4*>(p + 16);   // 8 bf16, K+16
    return f.v;
}

// B-matrix 32x16 bf16 fragment where B[k,n] = W[n0+n, k0+k] (W row-major,
// i.e. the W @ ... ^T operand). ISA layout:
//   lanes 0-15  : column n = lane,    v0-3 = K 0..7,  v4-7 = K 16..23
//   lanes 16-31 : column n = lane-16, v0-3 = K 8..15, v4-7 = K 24..31
__device__ inline v16bf load_bt_frag(const __bf16* W, int ldw, int n0, int k0, int lane) {
    int n  = lane & 15;
    int hf = (lane >> 4) & 1;
    const __bf16* p = W + (size_t)(n0 + n) * ldw + k0 + hf * 8;
    FragU f;
    f.q[0] = *reinterpret_cast<const uint4*>(p);
    f.q[1] = *reinterpret_cast<const uint4*>(p + 16);
    return f.v;
}

__device__ inline float sigm(float v) { return 1.0f / (1.0f + __expf(-v)); }

// ------------------------------ prep kernels -------------------------------

__global__ void cvt_f32_bf16(const float* __restrict__ src, __bf16* __restrict__ dst, int n) {
    for (int i = blockIdx.x * blockDim.x + threadIdx.x; i < n; i += gridDim.x * blockDim.x)
        dst[i] = (__bf16)src[i];
}

// UT[r*rows + i] = U[i*rank + r]   (transpose-convert, rank == RNK)
__global__ void tcvt_f32_bf16(const float* __restrict__ U, __bf16* __restrict__ UT, int rows) {
    int n = rows * RNK;
    for (int t = blockIdx.x * blockDim.x + threadIdx.x; t < n; t += gridDim.x * blockDim.x) {
        int i = t / RNK;
        int r = t - i * RNK;
        UT[(size_t)r * rows + i] = (__bf16)U[(size_t)i * RNK + r];
    }
}

// corr[g*dim + i] = sum_r U[i,r] * V[g*H + i, r]
__global__ void corr_kernel(const float* __restrict__ U, const float* __restrict__ V,
                            float* __restrict__ corr, int dim) {
    int n = 4 * dim;
    for (int t = blockIdx.x * blockDim.x + threadIdx.x; t < n; t += gridDim.x * blockDim.x) {
        int g = t / dim;
        int i = t - g * dim;
        const float* u = U + (size_t)i * RNK;
        const float* v = V + (size_t)(g * HDIM + i) * RNK;
        float s = 0.f;
        #pragma unroll 8
        for (int r = 0; r < RNK; ++r) s += u[r] * v[r];
        corr[t] = s;
    }
}

// ------------------------------ phase 1 ------------------------------------
// T[B x RNK] = A[B x K] @ U[K x RNK], UT is [RNK x K] row-major (bf16).
// One 16x16 output tile per wave; tiles = (B/16) * (RNK/16) = 2048.
__global__ void lowrank_T(const __bf16* __restrict__ A, int lda, int K,
                          const __bf16* __restrict__ UT, __bf16* __restrict__ T) {
    int lane = threadIdx.x & 31;
    int wave = threadIdx.x >> 5;
    int tile = blockIdx.x * (blockDim.x >> 5) + wave;
    int tm = tile >> 3;            // B/16 row tiles
    int tn = tile & 7;             // RNK/16 = 8 col tiles
    int row0 = tm * 16, col0 = tn * 16;

    v8f acc = {};
    for (int k = 0; k < K; k += 32) {
        v16bf a = load_a_frag(A, lda, row0, k, lane);
        v16bf b = load_bt_frag(UT, K, col0, k, lane);
        acc = __builtin_amdgcn_wmma_f32_16x16x32_bf16(false, a, false, b,
                                                      (short)0, acc, false, false);
    }
    // C layout: lane n / n+16 holds column n; v r holds row (hf*8 + r)
    int n  = lane & 15;
    int hf = (lane >> 4) & 1;
    #pragma unroll
    for (int r = 0; r < 8; ++r) {
        int row = row0 + hf * 8 + r;
        T[(size_t)row * RNK + col0 + n] = (__bf16)acc[r];
    }
}

// ------------------------------ phase 2 ------------------------------------
// For each 16(batch) x 16(hidden) tile: accumulate all 4 gate pre-activations
// from Tx@Vx^T + Th@Vh^T, then fused LSTM epilogue. tiles = 256*64 = 16384.
__global__ void vmlmf_phase2(const __bf16* __restrict__ Tx, const __bf16* __restrict__ Th,
                             const __bf16* __restrict__ Vxb, const __bf16* __restrict__ Vhb,
                             const float* __restrict__ x, const float* __restrict__ h,
                             const float* __restrict__ c,
                             const float* __restrict__ corrx, const float* __restrict__ corrh,
                             const float* __restrict__ bx, const float* __restrict__ bh,
                             const float* __restrict__ diax, const float* __restrict__ diah,
                             float* __restrict__ out) {
    int lane = threadIdx.x & 31;
    int wave = threadIdx.x >> 5;
    int tile = blockIdx.x * (blockDim.x >> 5) + wave;
    int tm = tile >> 6;            // B/16 = 256 row tiles
    int tn = tile & 63;            // H/16 = 64 col tiles
    int row0 = tm * 16, col0 = tn * 16;

    v8f acc0 = {}, acc1 = {}, acc2 = {}, acc3 = {};

    // x low-rank contribution
    #pragma unroll
    for (int k = 0; k < RNK; k += 32) {
        v16bf a  = load_a_frag(Tx, RNK, row0, k, lane);
        v16bf b0 = load_bt_frag(Vxb, RNK, 0 * HDIM + col0, k, lane);
        v16bf b1 = load_bt_frag(Vxb, RNK, 1 * HDIM + col0, k, lane);
        v16bf b2 = load_bt_frag(Vxb, RNK, 2 * HDIM + col0, k, lane);
        v16bf b3 = load_bt_frag(Vxb, RNK, 3 * HDIM + col0, k, lane);
        acc0 = __builtin_amdgcn_wmma_f32_16x16x32_bf16(false, a, false, b0, (short)0, acc0, false, false);
        acc1 = __builtin_amdgcn_wmma_f32_16x16x32_bf16(false, a, false, b1, (short)0, acc1, false, false);
        acc2 = __builtin_amdgcn_wmma_f32_16x16x32_bf16(false, a, false, b2, (short)0, acc2, false, false);
        acc3 = __builtin_amdgcn_wmma_f32_16x16x32_bf16(false, a, false, b3, (short)0, acc3, false, false);
    }
    // h low-rank contribution
    #pragma unroll
    for (int k = 0; k < RNK; k += 32) {
        v16bf a  = load_a_frag(Th, RNK, row0, k, lane);
        v16bf b0 = load_bt_frag(Vhb, RNK, 0 * HDIM + col0, k, lane);
        v16bf b1 = load_bt_frag(Vhb, RNK, 1 * HDIM + col0, k, lane);
        v16bf b2 = load_bt_frag(Vhb, RNK, 2 * HDIM + col0, k, lane);
        v16bf b3 = load_bt_frag(Vhb, RNK, 3 * HDIM + col0, k, lane);
        acc0 = __builtin_amdgcn_wmma_f32_16x16x32_bf16(false, a, false, b0, (short)0, acc0, false, false);
        acc1 = __builtin_amdgcn_wmma_f32_16x16x32_bf16(false, a, false, b1, (short)0, acc1, false, false);
        acc2 = __builtin_amdgcn_wmma_f32_16x16x32_bf16(false, a, false, b2, (short)0, acc2, false, false);
        acc3 = __builtin_amdgcn_wmma_f32_16x16x32_bf16(false, a, false, b3, (short)0, acc3, false, false);
    }

    // ----- fused epilogue -----
    int n  = lane & 15;
    int hf = (lane >> 4) & 1;
    int col = col0 + n;
    bool inI = (col < IDIM);

    float dxv = inI ? diax[col] : 0.f;
    float dhv = diah[col];
    float cx0 = inI ? corrx[0 * IDIM + col] : 0.f;
    float cx1 = inI ? corrx[1 * IDIM + col] : 0.f;
    float cx2 = inI ? corrx[2 * IDIM + col] : 0.f;
    float cx3 = inI ? corrx[3 * IDIM + col] : 0.f;
    float ch0 = corrh[0 * HDIM + col];
    float ch1 = corrh[1 * HDIM + col];
    float ch2 = corrh[2 * HDIM + col];
    float ch3 = corrh[3 * HDIM + col];
    float bb0 = bx[0 * HDIM + col] + bh[0 * HDIM + col];
    float bb1 = bx[1 * HDIM + col] + bh[1 * HDIM + col];
    float bb2 = bx[2 * HDIM + col] + bh[2 * HDIM + col];
    float bb3 = bx[3 * HDIM + col] + bh[3 * HDIM + col];

    #pragma unroll
    for (int r = 0; r < 8; ++r) {
        int row = row0 + hf * 8 + r;
        float xv = inI ? x[(size_t)row * IDIM + col] : 0.f;
        float hv = h[(size_t)row * HDIM + col];
        float cv = c[(size_t)row * HDIM + col];
        float add = dxv * xv + dhv * hv;

        float pi = acc0[r] - xv * cx0 - hv * ch0 + bb0 + add;
        float pf = acc1[r] - xv * cx1 - hv * ch1 + bb1 + add;
        float po = acc2[r] - xv * cx2 - hv * ch2 + bb2 + add;
        float pn = acc3[r] - xv * cx3 - hv * ch3 + bb3 + add;

        float ig = sigm(pi);
        float fg = sigm(pf);
        float og = sigm(po);
        float ng = tanhf(pn);

        float cn = fg * cv + ig * ng;
        float hn = og * tanhf(cn);

        size_t o = (size_t)row * HDIM + col;
        out[o] = hn;                                  // h_next
        out[(size_t)BDIM * HDIM + o] = cn;            // c_next
    }
}

// ------------------------------ launcher -----------------------------------

extern "C" void kernel_launch(void* const* d_in, const int* in_sizes, int n_in,
                              void* d_out, int out_size, void* d_ws, size_t ws_size,
                              hipStream_t stream) {
    (void)in_sizes; (void)n_in; (void)out_size; (void)ws_size;

    const float* x    = (const float*)d_in[0];
    const float* h    = (const float*)d_in[1];
    const float* c    = (const float*)d_in[2];
    const float* U_x  = (const float*)d_in[3];
    const float* U_h  = (const float*)d_in[4];
    const float* V_x  = (const float*)d_in[5];
    const float* V_h  = (const float*)d_in[6];
    const float* b_x  = (const float*)d_in[7];
    const float* b_h  = (const float*)d_in[8];
    const float* diax = (const float*)d_in[9];
    const float* diah = (const float*)d_in[10];
    float* out = (float*)d_out;

    // workspace carve-up (bf16 units; all sizes are multiples of 128 elems)
    __bf16* w   = (__bf16*)d_ws;
    __bf16* xb  = w;  w += (size_t)BDIM * IDIM;        // 4 MB
    __bf16* hb  = w;  w += (size_t)BDIM * HDIM;        // 8 MB
    __bf16* Vxb = w;  w += (size_t)4 * HDIM * RNK;     // 1 MB
    __bf16* Vhb = w;  w += (size_t)4 * HDIM * RNK;     // 1 MB
    __bf16* UxT = w;  w += (size_t)RNK * IDIM;         // 128 KB
    __bf16* UhT = w;  w += (size_t)RNK * HDIM;         // 256 KB
    __bf16* Tx  = w;  w += (size_t)BDIM * RNK;         // 1 MB
    __bf16* Th  = w;  w += (size_t)BDIM * RNK;         // 1 MB
    float*  corrx = (float*)w;
    float*  corrh = corrx + 4 * IDIM;

    // phase 0: conversions + small reductions
    cvt_f32_bf16<<<2048, 256, 0, stream>>>(x,   xb,  BDIM * IDIM);
    cvt_f32_bf16<<<2048, 256, 0, stream>>>(h,   hb,  BDIM * HDIM);
    cvt_f32_bf16<<<1024, 256, 0, stream>>>(V_x, Vxb, 4 * HDIM * RNK);
    cvt_f32_bf16<<<1024, 256, 0, stream>>>(V_h, Vhb, 4 * HDIM * RNK);
    tcvt_f32_bf16<<<256, 256, 0, stream>>>(U_x, UxT, IDIM);
    tcvt_f32_bf16<<<512, 256, 0, stream>>>(U_h, UhT, HDIM);
    corr_kernel<<<(4 * IDIM + 255) / 256, 256, 0, stream>>>(U_x, V_x, corrx, IDIM);
    corr_kernel<<<(4 * HDIM + 255) / 256, 256, 0, stream>>>(U_h, V_h, corrh, HDIM);

    // phase 1: Tx = x@U_x, Th = h@U_h  (2048 tiles each, 4 waves/block)
    lowrank_T<<<512, 128, 0, stream>>>(xb, IDIM, IDIM, UxT, Tx);
    lowrank_T<<<512, 128, 0, stream>>>(hb, HDIM, HDIM, UhT, Th);

    // phase 2: fused big GEMMs + LSTM epilogue (16384 tiles, 4 waves/block)
    vmlmf_phase2<<<4096, 128, 0, stream>>>(Tx, Th, Vxb, Vhb, x, h, c,
                                           corrx, corrh, b_x, b_h, diax, diah, out);
}
